// GNNSubstructures_21002390077869
// MI455X (gfx1250) — compile-verified, matrix operand order checked by reading, not run on hardware
//
#include <hip/hip_runtime.h>
#include <hip/hip_bf16.h>

typedef __attribute__((ext_vector_type(16))) __bf16 v16bf;
typedef __attribute__((ext_vector_type(8)))  float  v8f;

#define LDS_WAIT() asm volatile("s_wait_dscnt 0" ::: "memory")

// Fragment-major layout for 16x32 bf16 WMMA operands (A and B share it):
//   flat[(frag*32 + lane)*16 + elem], lane's 16 bf16 contiguous (32 bytes).
// Inverse of the ISA 7.12.2 mapping: for element (m_or_n, k):
//   lane = ((k>>3)&1)*16 + m_or_n ; elem = ((k>>4)&1)*8 + (k&7)
__device__ __forceinline__ v16bf load_frag(const __bf16* base, int fragIdx, int lane) {
    const uint4* p = (const uint4*)(base + (size_t)(fragIdx * 32 + lane) * 16);
    union { uint4 q[2]; v16bf v; } u;
    u.q[0] = p[0];
    u.q[1] = p[1];
    return u.v;
}

// Swizzle a K x 64 fp32 weight matrix into bf16 fragment-major order.
__global__ void swizzle_weight_kernel(const float* __restrict__ W, __bf16* __restrict__ dst,
                                      int K) {
    int i = blockIdx.x * blockDim.x + threadIdx.x;
    if (i >= K * 64) return;
    int k = i >> 6, n = i & 63;
    int kb = k >> 5, kl = k & 31;
    int nt = n >> 4, n15 = n & 15;
    int lane = ((kl >> 3) & 1) * 16 + n15;
    int elem = ((kl >> 4) & 1) * 8 + (kl & 7);
    dst[(size_t)((kb * 4 + nt) * 32 + lane) * 16 + elem] = (__bf16)W[i];
}

__global__ void zero_kernel(float* __restrict__ p, int n) {
    int i = blockIdx.x * blockDim.x + threadIdx.x;
    if (i < n) p[i] = 0.0f;
}

// h0 = x @ Wn + bn  (N x 64), ids = identifiers @ Wid + bid (N x 16)
__global__ void node_encode_kernel(const float* __restrict__ x, const float* __restrict__ idsin,
                                   const float* __restrict__ Wn, const float* __restrict__ bn,
                                   const float* __restrict__ Wid, const float* __restrict__ bid,
                                   float* __restrict__ h0f, __bf16* __restrict__ h0b,
                                   __bf16* __restrict__ idsb, int N) {
    int n = blockIdx.x * blockDim.x + threadIdx.x;
    if (n >= N) return;
    float xi[32];
#pragma unroll
    for (int k = 0; k < 32; ++k) xi[k] = x[(size_t)n * 32 + k];
    for (int j = 0; j < 64; ++j) {
        float acc = bn[j];
#pragma unroll
        for (int k = 0; k < 32; ++k) acc += xi[k] * Wn[k * 64 + j];
        h0f[(size_t)n * 64 + j] = acc;
        h0b[(size_t)n * 64 + j] = (__bf16)acc;
    }
    float ii[8];
#pragma unroll
    for (int k = 0; k < 8; ++k) ii[k] = idsin[(size_t)n * 8 + k];
    for (int j = 0; j < 16; ++j) {
        float acc = bid[j];
#pragma unroll
        for (int k = 0; k < 8; ++k) acc += ii[k] * Wid[k * 16 + j];
        idsb[(size_t)n * 16 + j] = (__bf16)acc;
    }
}

// Per-edge message MLP:  m = relu([feat] @ W1 + b1) @ W2 + b2, atomically
// scattered into agg[row].  One wave handles a 16-edge tile.
template <int K_IN, bool HAS_IDS>
__global__ __launch_bounds__(128) void edge_msg_kernel(
    const int* __restrict__ rowp, const int* __restrict__ colp,
    const __bf16* __restrict__ hb, const __bf16* __restrict__ idsb,
    const __bf16* __restrict__ W1b, const float* __restrict__ b1,
    const __bf16* __restrict__ W2b, const float* __restrict__ b2,
    float* __restrict__ agg, int nTiles, int E) {
    constexpr int WAVES = 4;
    constexpr int KB1 = K_IN / 32;
    constexpr int CPE = K_IN / 8;       // 16-byte chunks per edge feature row
    constexpr int CH  = 16 * CPE / 32;  // staging iterations per lane
    __shared__ __bf16 sW1[K_IN * 64];
    __shared__ __bf16 sA[WAVES][16 * K_IN];
    __shared__ __bf16 sH[WAVES][16 * 64];

    const int tid = threadIdx.x;
    {  // coalesced copy of swizzled W1 into LDS
        const uint4* src = (const uint4*)W1b;
        uint4* dst = (uint4*)sW1;
        for (int i = tid; i < K_IN * 64 / 8; i += 128) dst[i] = src[i];
    }
    __syncthreads();

    const int lane = tid & 31;
    const int wave = tid >> 5;
    const int half = lane >> 4;
    const int n15 = lane & 15;
    __bf16* A = sA[wave];
    __bf16* H = sH[wave];

    // W2 fragments live in registers for the whole kernel (swizzled global)
    v16bf w2f[8];
#pragma unroll
    for (int f = 0; f < 8; ++f) w2f[f] = load_frag(W2b, f, lane);

    float bj1[4], bj2[4];
#pragma unroll
    for (int nt = 0; nt < 4; ++nt) {
        bj1[nt] = b1[nt * 16 + n15];
        bj2[nt] = b2[nt * 16 + n15];
    }

    for (int tile = blockIdx.x * WAVES + wave; tile < nTiles; tile += (int)gridDim.x * WAVES) {
        const int e0 = tile * 16;
        // gather concat features directly into fragment-major LDS (b128 ops)
#pragma unroll
        for (int i = 0; i < CH; ++i) {
            const int chunk = lane + 32 * i;
            const int e = chunk / CPE;
            const int kc = chunk - e * CPE;
            const int k = kc * 8;
            int ee = e0 + e;
            if (ee >= E) ee = E - 1;
            uint4 val;
            if (k < 64)
                val = *(const uint4*)(hb + (size_t)rowp[ee] * 64 + k);
            else if (k < 128)
                val = *(const uint4*)(hb + (size_t)colp[ee] * 64 + (k - 64));
            else if (HAS_IDS) {
                if (k < 144)
                    val = *(const uint4*)(idsb + (size_t)rowp[ee] * 16 + (k - 128));
                else
                    val = *(const uint4*)(idsb + (size_t)colp[ee] * 16 + (k - 144));
            } else
                val = make_uint4(0u, 0u, 0u, 0u);
            const int kb = k >> 5, kl = k & 31;
            const int laneA = ((kl >> 3) & 1) * 16 + e;
            const int elemB = ((kl >> 4) & 1) * 8;
            *(uint4*)(A + kb * 512 + laneA * 16 + elemB) = val;
        }
        LDS_WAIT();

        v16bf af[KB1];
#pragma unroll
        for (int kb = 0; kb < KB1; ++kb) af[kb] = load_frag(A, kb, lane);

        // hidden = relu(A @ W1 + b1), written in fragment-major order
#pragma unroll
        for (int nt = 0; nt < 4; ++nt) {
            v8f c = {};
#pragma unroll
            for (int kb = 0; kb < KB1; ++kb) {
                v16bf b = load_frag(sW1, kb * 4 + nt, lane);
                c = __builtin_amdgcn_wmma_f32_16x16x32_bf16(false, af[kb], false, b,
                                                            (short)0, c, false, false);
            }
#pragma unroll
            for (int r = 0; r < 8; ++r) {
                float v = c[r] + bj1[nt];
                v = fmaxf(v, 0.0f);
                const int laneH = ((n15 >> 3) & 1) * 16 + (r + 8 * half);
                const int elem = (nt & 1) * 8 + (n15 & 7);
                H[(nt >> 1) * 512 + laneH * 16 + elem] = (__bf16)v;
            }
        }
        LDS_WAIT();

        v16bf a2[2];
        a2[0] = load_frag(H, 0, lane);
        a2[1] = load_frag(H, 1, lane);

        int dst[8];
#pragma unroll
        for (int r = 0; r < 8; ++r) {
            int ee = e0 + r + 8 * half;
            if (ee >= E) ee = E - 1;
            dst[r] = rowp[ee];
        }

        // msg = hidden @ W2 + b2; atomic scatter-add into agg[row]
#pragma unroll
        for (int nt = 0; nt < 4; ++nt) {
            v8f c = {};
#pragma unroll
            for (int kb = 0; kb < 2; ++kb)
                c = __builtin_amdgcn_wmma_f32_16x16x32_bf16(false, a2[kb], false,
                                                            w2f[kb * 4 + nt], (short)0, c,
                                                            false, false);
            const int nc = nt * 16 + n15;
#pragma unroll
            for (int r = 0; r < 8; ++r)
                atomicAdd(&agg[(size_t)dst[r] * 64 + nc], c[r] + bj2[nt]);
        }
    }
}

// Node update: out = relu( relu([h|agg] @ W1 + b1) @ W2 + b2 ), 16-node tiles.
template <bool STORE_BF16>
__global__ __launch_bounds__(128) void update_kernel(
    const __bf16* __restrict__ hb, const float* __restrict__ aggf,
    const __bf16* __restrict__ W1b, const float* __restrict__ b1,
    const __bf16* __restrict__ W2b, const float* __restrict__ b2,
    float* __restrict__ outf, __bf16* __restrict__ outb, int nTiles, int N) {
    constexpr int WAVES = 4;
    constexpr int K_IN = 128;
    constexpr int CPE = K_IN / 8;
    constexpr int CH  = 16 * CPE / 32;
    __shared__ __bf16 sW1[K_IN * 64];
    __shared__ __bf16 sA[WAVES][16 * K_IN];
    __shared__ __bf16 sH[WAVES][16 * 64];

    const int tid = threadIdx.x;
    {
        const uint4* src = (const uint4*)W1b;
        uint4* dst = (uint4*)sW1;
        for (int i = tid; i < K_IN * 64 / 8; i += 128) dst[i] = src[i];
    }
    __syncthreads();

    const int lane = tid & 31;
    const int wave = tid >> 5;
    const int half = lane >> 4;
    const int n15 = lane & 15;
    __bf16* A = sA[wave];
    __bf16* H = sH[wave];

    v16bf w2f[8];
#pragma unroll
    for (int f = 0; f < 8; ++f) w2f[f] = load_frag(W2b, f, lane);

    float bj1[4], bj2[4];
#pragma unroll
    for (int nt = 0; nt < 4; ++nt) {
        bj1[nt] = b1[nt * 16 + n15];
        bj2[nt] = b2[nt * 16 + n15];
    }

    for (int tile = blockIdx.x * WAVES + wave; tile < nTiles; tile += (int)gridDim.x * WAVES) {
        const int n0 = tile * 16;
#pragma unroll
        for (int i = 0; i < CH; ++i) {
            const int chunk = lane + 32 * i;
            const int e = chunk / CPE;
            const int kc = chunk - e * CPE;
            const int k = kc * 8;
            int nn = n0 + e;
            if (nn >= N) nn = N - 1;
            uint4 val;
            if (k < 64)
                val = *(const uint4*)(hb + (size_t)nn * 64 + k);
            else {
                const float4* f4 = (const float4*)(aggf + (size_t)nn * 64 + (k - 64));
                float4 f0 = f4[0], f1 = f4[1];
                union { __bf16 h[8]; uint4 q; } u;
                u.h[0] = (__bf16)f0.x; u.h[1] = (__bf16)f0.y;
                u.h[2] = (__bf16)f0.z; u.h[3] = (__bf16)f0.w;
                u.h[4] = (__bf16)f1.x; u.h[5] = (__bf16)f1.y;
                u.h[6] = (__bf16)f1.z; u.h[7] = (__bf16)f1.w;
                val = u.q;
            }
            const int kb = k >> 5, kl = k & 31;
            const int laneA = ((kl >> 3) & 1) * 16 + e;
            const int elemB = ((kl >> 4) & 1) * 8;
            *(uint4*)(A + kb * 512 + laneA * 16 + elemB) = val;
        }
        LDS_WAIT();

        v16bf af[4];
#pragma unroll
        for (int kb = 0; kb < 4; ++kb) af[kb] = load_frag(A, kb, lane);

#pragma unroll
        for (int nt = 0; nt < 4; ++nt) {
            v8f c = {};
#pragma unroll
            for (int kb = 0; kb < 4; ++kb) {
                v16bf b = load_frag(sW1, kb * 4 + nt, lane);
                c = __builtin_amdgcn_wmma_f32_16x16x32_bf16(false, af[kb], false, b,
                                                            (short)0, c, false, false);
            }
#pragma unroll
            for (int r = 0; r < 8; ++r) {
                float v = c[r] + bj1[nt];
                v = fmaxf(v, 0.0f);
                const int laneH = ((n15 >> 3) & 1) * 16 + (r + 8 * half);
                const int elem = (nt & 1) * 8 + (n15 & 7);
                H[(nt >> 1) * 512 + laneH * 16 + elem] = (__bf16)v;
            }
        }
        LDS_WAIT();

        v16bf a2[2];
        a2[0] = load_frag(H, 0, lane);
        a2[1] = load_frag(H, 1, lane);

#pragma unroll
        for (int nt = 0; nt < 4; ++nt) {
            v8f c = {};
#pragma unroll
            for (int kb = 0; kb < 2; ++kb)
                c = __builtin_amdgcn_wmma_f32_16x16x32_bf16(false, a2[kb], false,
                                                            w2f[kb * 4 + nt], (short)0, c,
                                                            false, false);
            const int nc = nt * 16 + n15;
#pragma unroll
            for (int r = 0; r < 8; ++r) {
                int nn = n0 + r + 8 * half;
                if (nn >= N) nn = N - 1;
                float v = c[r] + bj2[nt];
                v = fmaxf(v, 0.0f);  // outer relu of update
                outf[(size_t)nn * 64 + nc] = v;
                if (STORE_BF16) outb[(size_t)nn * 64 + nc] = (__bf16)v;
            }
        }
    }
}

// Atomic segment-sum pooling of h0/h1/h2 into [3][G,64]
__global__ void pool_kernel(const float* __restrict__ h0, const float* __restrict__ h1,
                            const float* __restrict__ h2, const int* __restrict__ batch,
                            float* __restrict__ pool, int N, int G) {
    int t = blockIdx.x * blockDim.x + threadIdx.x;
    if (t >= N * 64) return;
    int n = t >> 6, c = t & 63;
    int g = batch[n];
    atomicAdd(&pool[(size_t)g * 64 + c], h0[t]);
    atomicAdd(&pool[(size_t)G * 64 + (size_t)g * 64 + c], h1[t]);
    atomicAdd(&pool[(size_t)2 * G * 64 + (size_t)g * 64 + c], h2[t]);
}

// pred = pool0 @ p0W + p0b + pool1 @ p1W + p1b + pool2 @ p2W + p2b
__global__ void project_kernel(const float* __restrict__ pool,
                               const float* __restrict__ p0W, const float* __restrict__ p0b,
                               const float* __restrict__ p1W, const float* __restrict__ p1b,
                               const float* __restrict__ p2W, const float* __restrict__ p2b,
                               float* __restrict__ out, int G) {
    int t = blockIdx.x * blockDim.x + threadIdx.x;
    if (t >= G * 128) return;
    int g = t >> 7, o = t & 127;
    const float* q0 = pool + (size_t)g * 64;
    const float* q1 = pool + (size_t)G * 64 + (size_t)g * 64;
    const float* q2 = pool + (size_t)2 * G * 64 + (size_t)g * 64;
    float acc = p0b[o] + p1b[o] + p2b[o];
    for (int k = 0; k < 64; ++k)
        acc += q0[k] * p0W[k * 128 + o] + q1[k] * p1W[k * 128 + o] + q2[k] * p2W[k * 128 + o];
    out[t] = acc;
}

extern "C" void kernel_launch(void* const* d_in, const int* in_sizes, int n_in,
                              void* d_out, int out_size, void* d_ws, size_t ws_size,
                              hipStream_t stream) {
    const float* x    = (const float*)d_in[0];
    /* d_in[1] degrees: unused by reference */
    const float* ids  = (const float*)d_in[2];
    const int*   eidx = (const int*)d_in[3];
    const int*   batch= (const int*)d_in[4];
    const float* Wn   = (const float*)d_in[5];
    const float* bn   = (const float*)d_in[6];
    const float* Wid  = (const float*)d_in[7];
    const float* bid  = (const float*)d_in[8];
    const float* m0W1 = (const float*)d_in[9];
    const float* m0b1 = (const float*)d_in[10];
    const float* m0W2 = (const float*)d_in[11];
    const float* m0b2 = (const float*)d_in[12];
    const float* u0W1 = (const float*)d_in[13];
    const float* u0b1 = (const float*)d_in[14];
    const float* u0W2 = (const float*)d_in[15];
    const float* u0b2 = (const float*)d_in[16];
    const float* m1W1 = (const float*)d_in[17];
    const float* m1b1 = (const float*)d_in[18];
    const float* m1W2 = (const float*)d_in[19];
    const float* m1b2 = (const float*)d_in[20];
    const float* u1W1 = (const float*)d_in[21];
    const float* u1b1 = (const float*)d_in[22];
    const float* u1W2 = (const float*)d_in[23];
    const float* u1b2 = (const float*)d_in[24];
    const float* p0W  = (const float*)d_in[25];
    const float* p0b  = (const float*)d_in[26];
    const float* p1W  = (const float*)d_in[27];
    const float* p1b  = (const float*)d_in[28];
    const float* p2W  = (const float*)d_in[29];
    const float* p2b  = (const float*)d_in[30];
    float* out = (float*)d_out;

    const int N = in_sizes[0] / 32;
    const int E = in_sizes[3] / 2;
    const int G = out_size / 128;

    char* ws = (char*)d_ws;
    size_t off = 0;
    auto take = [&](size_t bytes) {
        char* p = ws + off;
        off = (off + bytes + 255) & ~(size_t)255;
        return p;
    };
    float*  h0f  = (float*) take((size_t)N * 64 * 4);
    __bf16* h0b  = (__bf16*)take((size_t)N * 64 * 2);
    __bf16* idsb = (__bf16*)take((size_t)N * 16 * 2);
    float*  agg  = (float*) take((size_t)N * 64 * 4);
    float*  h1f  = (float*) take((size_t)N * 64 * 4);
    __bf16* h1b  = (__bf16*)take((size_t)N * 64 * 2);
    float*  h2f  = (float*) take((size_t)N * 64 * 4);
    float*  pool = (float*) take((size_t)3 * G * 64 * 4);
    __bf16* wb   = (__bf16*)take((size_t)51200 * 2);

    __bf16* m0W1b = wb;          // 160*64 (swizzled fragment-major)
    __bf16* m0W2b = wb + 10240;  // 64*64
    __bf16* u0W1b = wb + 14336;  // 128*64
    __bf16* u0W2b = wb + 22528;  // 64*64
    __bf16* m1W1b = wb + 26624;  // 128*64
    __bf16* m1W2b = wb + 34816;  // 64*64
    __bf16* u1W1b = wb + 38912;  // 128*64
    __bf16* u1W2b = wb + 47104;  // 64*64

    auto swz = [&](const float* s, __bf16* d, int K) {
        swizzle_weight_kernel<<<(K * 64 + 255) / 256, 256, 0, stream>>>(s, d, K);
    };
    swz(m0W1, m0W1b, 160);
    swz(m0W2, m0W2b, 64);
    swz(u0W1, u0W1b, 128);
    swz(u0W2, u0W2b, 64);
    swz(m1W1, m1W1b, 128);
    swz(m1W2, m1W2b, 64);
    swz(u1W1, u1W1b, 128);
    swz(u1W2, u1W2b, 64);

    node_encode_kernel<<<(N + 127) / 128, 128, 0, stream>>>(x, ids, Wn, bn, Wid, bid,
                                                            h0f, h0b, idsb, N);

    zero_kernel<<<(N * 64 + 255) / 256, 256, 0, stream>>>(agg, N * 64);
    zero_kernel<<<(3 * G * 64 + 255) / 256, 256, 0, stream>>>(pool, 3 * G * 64);

    const int eTiles = (E + 15) / 16;
    const int nTiles = (N + 15) / 16;
    const dim3 eGrid((eTiles + 3) / 4);
    const dim3 nGrid((nTiles + 3) / 4);

    // layer 0: GSN message (K = 2*64 + 2*16 = 160, with ids) + update
    edge_msg_kernel<160, true><<<eGrid, 128, 0, stream>>>(
        eidx, eidx + E, h0b, idsb, m0W1b, m0b1, m0W2b, m0b2, agg, eTiles, E);
    update_kernel<true><<<nGrid, 128, 0, stream>>>(
        h0b, agg, u0W1b, u0b1, u0W2b, u0b2, h1f, h1b, nTiles, N);

    // layer 1: MPNN message (K = 2*64 = 128, no ids) + update
    zero_kernel<<<(N * 64 + 255) / 256, 256, 0, stream>>>(agg, N * 64);
    edge_msg_kernel<128, false><<<eGrid, 128, 0, stream>>>(
        eidx, eidx + E, h1b, (const __bf16*)nullptr, m1W1b, m1b1, m1W2b, m1b2, agg, eTiles, E);
    update_kernel<false><<<nGrid, 128, 0, stream>>>(
        h1b, agg, u1W1b, u1b1, u1W2b, u1b2, h2f, (__bf16*)nullptr, nTiles, N);

    // jumping-knowledge readout
    pool_kernel<<<(N * 64 + 255) / 256, 256, 0, stream>>>(h0f, h1f, h2f, batch, pool, N, G);
    project_kernel<<<(G * 128 + 255) / 256, 256, 0, stream>>>(pool, p0W, p0b, p1W, p1b,
                                                              p2W, p2b, out, G);
}
